// DenoisingModel_75161927680524
// MI455X (gfx1250) — matrix operand order repaired and checked by reading, not compiled, not run on hardware
//
#include <hip/hip_runtime.h>
#include <math.h>
#include <stdint.h>

// ---------------- model constants ----------------
#define B_SZ   16
#define HH     32
#define WW     32
#define NPIX   1024              // H*W
#define MROWS  (B_SZ * NPIX)     // 16384 rows for all (B,N,C) GEMMs
#define MT     (MROWS / 16)      // 1024 M-tiles
#define KP     128               // padded channel stride (covers C<=97, mult of 32)
#define KNN    8
#define EPSBN  1e-5f
#define SLOPE  0.05f

typedef __attribute__((ext_vector_type(16))) _Float16 v16h;
typedef __attribute__((ext_vector_type(8)))  float    v8f;

__device__ __forceinline__ float leakyf(float v) { return v >= 0.f ? v : SLOPE * v; }

// ---------------- Tensor Data Mover support (gated) ----------------
#if defined(__has_builtin)
#if __has_builtin(__builtin_amdgcn_tensor_load_to_lds) && __has_builtin(__builtin_amdgcn_s_wait_tensorcnt)
#define USE_TDM 1
#endif
#endif
#ifndef USE_TDM
#define USE_TDM 0
#endif

#if USE_TDM
typedef unsigned int u32x4 __attribute__((ext_vector_type(4)));
typedef int          i32x4 __attribute__((ext_vector_type(4)));
typedef int          i32x8 __attribute__((ext_vector_type(8)));

// DMA-copy 1024 contiguous bytes (512 f16) from global to LDS byte offset lds_off.
// D# built per CDNA5 ISA ch.8: group0 = {count=1 | lds_addr | global_addr | type=2},
// group1 = {data_size=2B, tensor_dim0=1024, tensor_dim1=1, tile_dim0=512, tile_dim1=1}.
__device__ __forceinline__ void tdm_copy_1k(const void* gsrc, unsigned lds_off)
{
    unsigned long long ga = (unsigned long long)(uintptr_t)gsrc;
    u32x4 g0 = { 1u,                                        // count=1, user mode
                 lds_off,                                   // LDS byte address
                 (unsigned)(ga & 0xFFFFFFFFu),
                 (unsigned)((ga >> 32) & 0x01FFFFFFu) | (2u << 30) };  // type=2 (image)
    i32x8 g1 = { (int)(1u << 16),                           // data_size code 1 = 2 bytes
                 (int)((1024u & 0xFFFFu) << 16),            // tensor_dim0 = 1024 (lo16 -> bits 63:48)
                 (int)((1u) << 16),                         // tensor_dim0 hi=0 | tensor_dim1 lo16 = 1
                 (int)((512u) << 16),                       // tensor_dim1 hi=0 | tile_dim0 = 512
                 (int)(1u),                                 // tile_dim1 = 1, tile_dim2 = 0
                 (int)(1024u),                              // tensor_dim0_stride lo
                 0, 0 };
    i32x4 gz = { 0, 0, 0, 0 };
#if __clang_major__ >= 23
    i32x8 gz8 = { 0, 0, 0, 0, 0, 0, 0, 0 };
    __builtin_amdgcn_tensor_load_to_lds(g0, g1, gz, gz, gz8, 0);
#else
    __builtin_amdgcn_tensor_load_to_lds(g0, g1, gz, gz, 0);
#endif
}
#endif

// ---------------- direct conv2d (SAME, NCHW) + bias + optional leaky ----------------
__global__ void conv2d_k(const float* __restrict__ x, const float* __restrict__ w,
                         const float* __restrict__ bias, float* __restrict__ out,
                         int IC, int OC, int ks, int act)
{
    int idx = blockIdx.x * blockDim.x + threadIdx.x;
    int total = B_SZ * OC * NPIX;
    if (idx >= total) return;
    int n  = idx % NPIX;
    int t  = idx / NPIX;
    int oc = t % OC;
    int b  = t / OC;
    int h = n >> 5, wq = n & 31;
    int pad = ks >> 1;
    float acc = bias[oc];
    for (int ic = 0; ic < IC; ++ic) {
        const float* xp = x + ((size_t)b * IC + ic) * NPIX;
        const float* wp = w + ((size_t)oc * IC + ic) * ks * ks;
        for (int kh = 0; kh < ks; ++kh) {
            int ih = h + kh - pad;
            if (ih < 0 || ih >= HH) continue;
            for (int kw = 0; kw < ks; ++kw) {
                int iw = wq + kw - pad;
                if (iw < 0 || iw >= WW) continue;
                acc = fmaf(xp[ih * WW + iw], wp[kh * ks + kw], acc);
            }
        }
    }
    if (act) acc = leakyf(acc);
    out[idx] = acc;
}

// ---------------- batchnorm (NCHW layout) ----------------
__global__ void bn_stats_nchw_k(const float* __restrict__ x, float* __restrict__ mv, int C)
{
    int c = blockIdx.x;
    __shared__ float s1[256], s2[256];
    float a = 0.f, b2 = 0.f;
    for (int i = threadIdx.x; i < B_SZ * NPIX; i += 256) {
        int b = i / NPIX, n = i % NPIX;
        float v = x[((size_t)b * C + c) * NPIX + n];
        a += v; b2 += v * v;
    }
    s1[threadIdx.x] = a; s2[threadIdx.x] = b2;
    __syncthreads();
    for (int o = 128; o > 0; o >>= 1) {
        if (threadIdx.x < o) { s1[threadIdx.x] += s1[threadIdx.x + o]; s2[threadIdx.x] += s2[threadIdx.x + o]; }
        __syncthreads();
    }
    if (threadIdx.x == 0) {
        float m = s1[0] / (float)(B_SZ * NPIX);
        float v = s2[0] / (float)(B_SZ * NPIX) - m * m;
        mv[c] = m; mv[C + c] = v;
    }
}

__global__ void bn_apply_nchw_k(const float* __restrict__ x, const float* __restrict__ mv,
                                const float* __restrict__ g, const float* __restrict__ bb,
                                float* __restrict__ out, int C, int act)
{
    int idx = blockIdx.x * blockDim.x + threadIdx.x;
    int total = B_SZ * C * NPIX;
    if (idx >= total) return;
    int t = idx / NPIX;
    int c = t % C;
    float r = (x[idx] - mv[c]) * rsqrtf(mv[C + c] + EPSBN) * g[c] + bb[c];
    if (act) r = leakyf(r);
    out[idx] = r;
}

// ---------------- batchnorm ((B*N, KP) row-major layout) ----------------
__global__ void bn_stats_bnc_k(const float* __restrict__ f, float* __restrict__ mv, int C)
{
    int c = blockIdx.x;
    __shared__ float s1[256], s2[256];
    float a = 0.f, b2 = 0.f;
    for (int i = threadIdx.x; i < MROWS; i += 256) {
        float v = f[(size_t)i * KP + c];
        a += v; b2 += v * v;
    }
    s1[threadIdx.x] = a; s2[threadIdx.x] = b2;
    __syncthreads();
    for (int o = 128; o > 0; o >>= 1) {
        if (threadIdx.x < o) { s1[threadIdx.x] += s1[threadIdx.x + o]; s2[threadIdx.x] += s2[threadIdx.x + o]; }
        __syncthreads();
    }
    if (threadIdx.x == 0) {
        float m = s1[0] / (float)MROWS;
        float v = s2[0] / (float)MROWS - m * m;
        mv[c] = m; mv[C + c] = v;
    }
}

__global__ void bn_apply_bnc_k(const float* __restrict__ f, const float* __restrict__ mv,
                               const float* __restrict__ g, const float* __restrict__ bb,
                               float* __restrict__ out, int C, int act)
{
    int idx = blockIdx.x * blockDim.x + threadIdx.x;
    int total = MROWS * C;
    if (idx >= total) return;
    int r = idx / C, c = idx % C;
    float v = (f[(size_t)r * KP + c] - mv[c]) * rsqrtf(mv[C + c] + EPSBN) * g[c] + bb[c];
    if (act) v = leakyf(v);
    out[(size_t)r * KP + c] = v;
}

// ---------------- layout transforms ----------------
__global__ void pack_k(const float* __restrict__ x, float* __restrict__ f, int C)
{
    int idx = blockIdx.x * blockDim.x + threadIdx.x;
    int total = B_SZ * C * NPIX;
    if (idx >= total) return;
    int n = idx % NPIX;
    int t = idx / NPIX;
    int c = t % C;
    int b = t / C;
    f[((size_t)(b * NPIX + n)) * KP + c] = x[idx];
}

// zero pad columns [C, KP) so hot GEMM paths need no bounds checks
__global__ void zeropad_k(float* __restrict__ f, int C)
{
    int r = blockIdx.x;
    for (int c = C + (int)threadIdx.x; c < KP; c += 32) f[(size_t)r * KP + c] = 0.f;
}

__global__ void unpack_k(const float* __restrict__ g, float* __restrict__ dst,
                         int Dsrc, int coff, int Cdst)
{
    int idx = blockIdx.x * blockDim.x + threadIdx.x;
    int total = B_SZ * Dsrc * NPIX;
    if (idx >= total) return;
    int n = idx % NPIX;
    int t = idx / NPIX;
    int d = t % Dsrc;
    int b = t / Dsrc;
    dst[((size_t)b * Cdst + coff + d) * NPIX + n] = g[((size_t)(b * NPIX + n)) * KP + d];
}

__global__ void rowsq_k(const float* __restrict__ f, float* __restrict__ sq, int C)
{
    int r = blockIdx.x * blockDim.x + threadIdx.x;
    if (r >= MROWS) return;
    float s = 0.f;
    for (int c = 0; c < C; ++c) { float v = f[(size_t)r * KP + c]; s = fmaf(v, v, s); }
    sq[r] = s;
}

__global__ void add2_k(const float* a, const float* b, float* o, int total)
{
    int i = blockIdx.x * blockDim.x + threadIdx.x;
    if (i < total) o[i] = a[i] + b[i];
}
__global__ void add3_k(const float* a, const float* b, const float* c, float* o, int total)
{
    int i = blockIdx.x * blockDim.x + threadIdx.x;
    if (i < total) o[i] = a[i] + b[i] + c[i];
}
__global__ void final_k(const float* __restrict__ g, const float* __restrict__ x, float* __restrict__ o)
{
    int i = blockIdx.x * blockDim.x + threadIdx.x;
    if (i < MROWS) o[i] = g[(size_t)i * KP] + x[i];
}

// ---------------- operand pre-swizzle into WMMA register layout ----------------
// A-operand layout (16x32 f16): lane L holds row (L&15); halves j -> K=kb+j,
// kb = 16 if L>=16. Output: [tile][ktile][lane][16 halves] contiguous (32B per lane).
__global__ void swzA_k(const float* __restrict__ f, _Float16* __restrict__ Az, int KT)
{
    int idx = blockIdx.x * blockDim.x + threadIdx.x;
    int total = MT * KT * 32;
    if (idx >= total) return;
    int L  = idx & 31;
    int t  = idx >> 5;
    int kt = t % KT;
    int mt = t / KT;
    int row = mt * 16 + (L & 15);
    const float* src = f + (size_t)row * KP + kt * 32 + ((L & 16) ? 16 : 0);
    _Float16* dst = Az + (size_t)idx * 16;
#pragma unroll
    for (int j = 0; j < 16; ++j) dst[j] = (_Float16)src[j];
}

// B-operand layout (32x16 f16) for B = F^T (gram): lane L holds column (L&15) = feature row,
// halves j -> K = kb+j with kb = 16 if L>=16 (16 contiguous K values).
__global__ void swzB_k(const float* __restrict__ f, _Float16* __restrict__ Bz, int KT)
{
    int idx = blockIdx.x * blockDim.x + threadIdx.x;
    int total = MT * KT * 32;
    if (idx >= total) return;
    int L  = idx & 31;
    int t  = idx >> 5;
    int kt = t % KT;
    int mt = t / KT;
    int row = mt * 16 + (L & 15);
    const float* src = f + (size_t)row * KP + kt * 32 + ((L & 16) ? 16 : 0);
    _Float16* dst = Bz + (size_t)idx * 16;
#pragma unroll
    for (int j = 0; j < 16; ++j) dst[j] = (_Float16)src[j];
}

// Weight (C,D) row-major -> B-operand layout [ntile][ktile][lane][16], zero padded.
__global__ void swzW_k(const float* __restrict__ w, _Float16* __restrict__ Wz,
                       int C, int D, int KT, int NT)
{
    int idx = blockIdx.x * blockDim.x + threadIdx.x;
    int total = NT * KT * 32;
    if (idx >= total) return;
    int L  = idx & 31;
    int t  = idx >> 5;
    int kt = t % KT;
    int nt = t / KT;
    int n  = nt * 16 + (L & 15);
    int kb = kt * 32 + ((L & 16) ? 16 : 0);
    _Float16* dst = Wz + (size_t)idx * 16;
#pragma unroll
    for (int j = 0; j < 16; ++j) {
        int k = kb + j;
        dst[j] = (k < C && n < D) ? (_Float16)w[(size_t)k * D + n] : (_Float16)0.f;
    }
}

// ---------------- WMMA pairwise-distance (Gram) kernel ----------------
// One wave per 16x16 tile: D = sq_r + sq_c - 2 * F F^T. Panels staged LDS via TDM when available.
__global__ void gram_dist_k(const _Float16* __restrict__ Az, const _Float16* __restrict__ Bz,
                            const float* __restrict__ sq, float* __restrict__ Dm, int KT)
{
    const int L  = threadIdx.x;
    const int b  = blockIdx.z;
    const int rt = blockIdx.x, ct = blockIdx.y;
    const size_t at = (size_t)(b * 64 + rt) * KT;   // A tile row index (global m-tile)
    const size_t bt = (size_t)(b * 64 + ct) * KT;
    v8f acc = {};
#if USE_TDM
    __shared__ _Float16 lbuf[1024];                 // 2 x 512-half panels
    for (int kt = 0; kt < KT; ++kt) {
        __syncthreads();
        tdm_copy_1k(&Az[(at + kt) * 512], 0u);
        tdm_copy_1k(&Bz[(bt + kt) * 512], 1024u);
        __builtin_amdgcn_s_wait_tensorcnt(0);
        __syncthreads();
        v16h av = *(const v16h*)&lbuf[L * 16];
        v16h bv = *(const v16h*)&lbuf[512 + L * 16];
        acc = __builtin_amdgcn_wmma_f32_16x16x32_f16(false, av, false, bv, (short)0, acc, false, false);
    }
#else
    for (int kt = 0; kt < KT; ++kt) {
        v16h av = *(const v16h*)&Az[((at + kt) * 32 + L) * 16];
        v16h bv = *(const v16h*)&Bz[((bt + kt) * 32 + L) * 16];
        acc = __builtin_amdgcn_wmma_f32_16x16x32_f16(false, av, false, bv, (short)0, acc, false, false);
    }
#endif
    const size_t base = (size_t)b * NPIX;
    int nn = L & 15;
    float sc = sq[base + ct * 16 + nn];
#pragma unroll
    for (int r = 0; r < 8; ++r) {
        int mm = r + ((L & 16) ? 8 : 0);
        Dm[((size_t)b * NPIX + rt * 16 + mm) * NPIX + (ct * 16 + nn)] =
            sq[base + rt * 16 + mm] + sc - 2.f * acc[r];
    }
}

// ---------------- top-8 neighbor selection + mean-diff feature ----------------
__global__ void topk_mdiff_k(const float* __restrict__ Dm, const float* __restrict__ f,
                             float* __restrict__ md, int C)
{
    int row = blockIdx.x;               // < MROWS
    int b = row >> 10, n = row & 1023;
    const float* dr = Dm + ((size_t)b * NPIX + n) * NPIX;
    int L = threadIdx.x;
    int sel[KNN];
    for (int k = 0; k < KNN; ++k) {
        float best = 3.4e38f; int bi = 0x7fffffff;
        for (int j = L; j < NPIX; j += 32) {
            bool used = false;
            for (int p = 0; p < k; ++p) used |= (sel[p] == j);
            if (used) continue;
            float v = dr[j];
            if (v < best || (v == best && j < bi)) { best = v; bi = j; }
        }
        for (int off = 16; off > 0; off >>= 1) {
            float ov = __shfl_xor(best, off, 32);
            int   oi = __shfl_xor(bi,  off, 32);
            if (ov < best || (ov == best && oi < bi)) { best = ov; bi = oi; }
        }
        sel[k] = bi;
    }
    const size_t base = (size_t)b * NPIX;
    for (int c = L; c < C; c += 32) {
        float s = 0.f;
#pragma unroll
        for (int k = 0; k < KNN; ++k) s += f[(base + sel[k]) * KP + c];
        md[(size_t)row * KP + c] = f[(size_t)row * KP + c] - s * (1.f / KNN);
    }
}

// ---------------- WMMA graph-conv kernel (pure register path) ----------------
// y = mdiff @ Wd + f @ Ws + (bd + bs); all operands pre-swizzled, 32B b128 loads per lane.
__global__ void gconv_wmma_k(const _Float16* __restrict__ Az, const _Float16* __restrict__ Mz,
                             const _Float16* __restrict__ WzS, const _Float16* __restrict__ WzD,
                             const float* __restrict__ bS, const float* __restrict__ bD,
                             float* __restrict__ out, int KT, int D, int act)
{
    const int L  = threadIdx.x;
    const int mt = blockIdx.x, nt = blockIdx.y;
    v8f acc = {};
    for (int kt = 0; kt < KT; ++kt) {
        v16h av = *(const v16h*)&Az[(((size_t)mt * KT + kt) * 32 + L) * 16];
        v16h mv = *(const v16h*)&Mz[(((size_t)mt * KT + kt) * 32 + L) * 16];
        v16h bs = *(const v16h*)&WzS[(((size_t)nt * KT + kt) * 32 + L) * 16];
        v16h bd = *(const v16h*)&WzD[(((size_t)nt * KT + kt) * 32 + L) * 16];
        acc = __builtin_amdgcn_wmma_f32_16x16x32_f16(false, av, false, bs, (short)0, acc, false, false);
        acc = __builtin_amdgcn_wmma_f32_16x16x32_f16(false, mv, false, bd, (short)0, acc, false, false);
    }
    int nn = L & 15, gn = nt * 16 + nn;
    float bias = (gn < D) ? (bS[gn] + bD[gn]) : 0.f;
#pragma unroll
    for (int r = 0; r < 8; ++r) {
        int mm = mt * 16 + r + ((L & 16) ? 8 : 0);
        float v = acc[r] + bias;
        if (act == 1)      v = leakyf(v);
        else if (act == 2) v = 1.f / (1.f + __expf(-v));
        out[(size_t)mm * KP + gn] = v;
    }
}

// ============================ host orchestration ============================
struct Lin  { const float *b, *w; };
struct GCp  { Lin d, s; };
struct Convp{ const float *b, *w; };
struct BNp  { const float *b, *g; };

static inline int cdiv(int a, int b) { return (a + b - 1) / b; }

struct Ws {
    float *tA, *tB, *yb, *yhpf, *f0, *f1, *md, *sq, *mv, *Dm;
    _Float16 *Az, *Bz, *Mz, *WzS, *WzD;
};

static void h_conv(hipStream_t s, const float* x, Convp c, int IC, int OC, int ks, int act, float* out)
{
    int total = B_SZ * OC * NPIX;
    conv2d_k<<<cdiv(total, 256), 256, 0, s>>>(x, c.w, c.b, out, IC, OC, ks, act);
}
static void h_bn_nchw(hipStream_t s, const float* x, BNp p, int C, int act, float* mv, float* out)
{
    bn_stats_nchw_k<<<C, 256, 0, s>>>(x, mv, C);
    bn_apply_nchw_k<<<cdiv(B_SZ * C * NPIX, 256), 256, 0, s>>>(x, mv, p.g, p.b, out, C, act);
}
static void h_bn_bnc(hipStream_t s, const float* fin, BNp p, int C, int act, float* mv, float* out)
{
    bn_stats_bnc_k<<<C, 256, 0, s>>>(fin, mv, C);
    bn_apply_bnc_k<<<cdiv(MROWS * C, 256), 256, 0, s>>>(fin, mv, p.g, p.b, out, C, act);
    zeropad_k<<<MROWS, 32, 0, s>>>(out, C);
}
static void h_pack(hipStream_t s, const float* x, int C, float* f)
{
    pack_k<<<cdiv(B_SZ * C * NPIX, 256), 256, 0, s>>>(x, f, C);
    zeropad_k<<<MROWS, 32, 0, s>>>(f, C);
}
// graph build: distances (WMMA) -> top-8 -> mean-diff; also leaves swizzled mdiff in W.Mz
static void h_graph(hipStream_t s, const float* f, int C, Ws& W)
{
    int KT = cdiv(C, 32);
    rowsq_k<<<cdiv(MROWS, 256), 256, 0, s>>>(f, W.sq, C);
    swzA_k<<<cdiv(MT * KT * 32, 256), 256, 0, s>>>(f, W.Az, KT);
    swzB_k<<<cdiv(MT * KT * 32, 256), 256, 0, s>>>(f, W.Bz, KT);
    gram_dist_k<<<dim3(64, 64, B_SZ), 32, 0, s>>>(W.Az, W.Bz, W.sq, W.Dm, KT);
    topk_mdiff_k<<<MROWS, 32, 0, s>>>(W.Dm, f, W.md, C);
    zeropad_k<<<MROWS, 32, 0, s>>>(W.md, C);
    swzA_k<<<cdiv(MT * KT * 32, 256), 256, 0, s>>>(W.md, W.Mz, KT);
}
static void h_gconv(hipStream_t s, const float* f, GCp g, int C, int D, int act, float* out, Ws& W)
{
    int KT = cdiv(C, 32), NT = cdiv(D, 16);
    swzA_k<<<cdiv(MT * KT * 32, 256), 256, 0, s>>>(f, W.Az, KT);
    swzW_k<<<cdiv(NT * KT * 32, 256), 256, 0, s>>>(g.s.w, W.WzS, C, D, KT, NT);
    swzW_k<<<cdiv(NT * KT * 32, 256), 256, 0, s>>>(g.d.w, W.WzD, C, D, KT, NT);
    gconv_wmma_k<<<dim3(MT, NT), 32, 0, s>>>(W.Az, W.Mz, W.WzS, W.WzD, g.s.b, g.d.b, out, KT, D, act);
    zeropad_k<<<MROWS, 32, 0, s>>>(out, D);
}
static void h_ppb(hipStream_t s, const float* x, const Convp cv[3], GCp gc, int ks, Ws& W)
{
    h_conv(s, x,    cv[0], 1, 32, ks, 1, W.tA);
    h_conv(s, W.tA, cv[1], 32, 32, ks, 1, W.tB);
    h_conv(s, W.tB, cv[2], 32, 32, ks, 1, W.tA);
    h_pack(s, W.tA, 32, W.f0);
    h_graph(s, W.f0, 32, W);
    h_gconv(s, W.f0, gc, 32, 32, 1, W.f1, W);       // result in f1
}

extern "C" void kernel_launch(void* const* d_in, const int* in_sizes, int n_in,
                              void* d_out, int out_size, void* d_ws, size_t ws_size,
                              hipStream_t stream)
{
    (void)in_sizes; (void)n_in; (void)out_size; (void)ws_size;

    // ---- parameter extraction (jax pytree order: dict keys sorted, lists in order) ----
    int pi = 0;
    auto nx     = [&]() { return (const float*)d_in[pi++]; };
    auto rdBN   = [&](BNp& p)  { p.b = nx(); p.g = nx(); };
    auto rdConv = [&](Convp& c){ c.b = nx(); c.w = nx(); };
    auto rdLin  = [&](Lin& l)  { l.b = nx(); l.w = nx(); };
    auto rdGC   = [&](GCp& g)  { rdLin(g.d); rdLin(g.s); };

    BNp   hpf_bn;            Convp hpf_conv;            GCp hpf_gcs[3];
    BNp   lpf_bn_c[4];       BNp   lpf_bns[4][3];       Convp lpf_conv[4];  GCp lpf_gcs[4][3];
    BNp   post_bns[2];       GCp   post_gcs[3];
    Convp pre_convs[3][3];   GCp   pre_gc[3];
    GCp   roi_final;         GCp   roi_gcs[8];          Convp roi_ppb_convs[3]; GCp roi_ppb_gc;

    rdBN(hpf_bn); rdConv(hpf_conv);
    for (int i = 0; i < 3; ++i) rdGC(hpf_gcs[i]);
    for (int l = 0; l < 4; ++l) {
        rdBN(lpf_bn_c[l]);
        for (int i = 0; i < 3; ++i) rdBN(lpf_bns[l][i]);
        rdConv(lpf_conv[l]);
        for (int i = 0; i < 3; ++i) rdGC(lpf_gcs[l][i]);
    }
    rdBN(post_bns[0]); rdBN(post_bns[1]);
    for (int i = 0; i < 3; ++i) rdGC(post_gcs[i]);
    for (int j = 0; j < 3; ++j) {
        for (int i = 0; i < 3; ++i) rdConv(pre_convs[j][i]);
        rdGC(pre_gc[j]);
    }
    rdGC(roi_final);
    for (int i = 0; i < 8; ++i) rdGC(roi_gcs[i]);
    for (int i = 0; i < 3; ++i) rdConv(roi_ppb_convs[i]);
    rdGC(roi_ppb_gc);
    const float* x = nx();

    // ---- workspace carve (whole working set ~120MB fits in MI455X 192MB L2) ----
    uintptr_t wp = (uintptr_t)d_ws;
    auto carve = [&](size_t bytes) -> void* {
        void* p = (void*)wp; wp += (bytes + 255) & ~(size_t)255; return p;
    };
    const size_t nchw97 = (size_t)B_SZ * 97 * NPIX * sizeof(float);
    const size_t frows  = (size_t)MROWS * KP * sizeof(float);
    const size_t swz    = (size_t)MT * 4 * 32 * 16 * sizeof(_Float16);   // KT<=4
    Ws W;
    W.tA   = (float*)carve(nchw97);
    W.tB   = (float*)carve(nchw97);
    W.yb   = (float*)carve(nchw97);
    W.yhpf = (float*)carve(nchw97);
    W.f0   = (float*)carve(frows);
    W.f1   = (float*)carve(frows);
    W.md   = (float*)carve(frows);
    W.sq   = (float*)carve((size_t)MROWS * sizeof(float));
    W.mv   = (float*)carve(2 * KP * sizeof(float));
    W.Az   = (_Float16*)carve(swz);
    W.Bz   = (_Float16*)carve(swz);
    W.Mz   = (_Float16*)carve(swz);
    W.WzS  = (_Float16*)carve((size_t)7 * 4 * 32 * 16 * sizeof(_Float16));
    W.WzD  = (_Float16*)carve((size_t)7 * 4 * 32 * 16 * sizeof(_Float16));
    W.Dm   = (float*)carve((size_t)B_SZ * NPIX * NPIX * sizeof(float));

    const int tot97 = B_SZ * 97 * NPIX;

    // ================= roi branch -> hits (channel 96 of y) =================
    h_ppb(stream, x, roi_ppb_convs, roi_ppb_gc, 7, W);
    float* cur = W.f1; float* alt = W.f0;
    for (int i = 0; i < 8; ++i) {
        if (i % 3 == 0) h_graph(stream, cur, 32, W);
        h_gconv(stream, cur, roi_gcs[i], 32, 32, 1, alt, W);
        float* t = cur; cur = alt; alt = t;
    }
    h_gconv(stream, cur, roi_final, 32, 1, 2, alt, W);               // sigmoid
    unpack_k<<<cdiv(B_SZ * NPIX, 256), 256, 0, stream>>>(alt, W.yb, 1, 96, 97);

    // ================= pre branches -> channels [0,96) of y =================
    const int ksv[3] = {5, 7, 9};
    for (int j = 0; j < 3; ++j) {
        h_ppb(stream, x, pre_convs[j], pre_gc[j], ksv[j], W);
        unpack_k<<<cdiv(B_SZ * 32 * NPIX, 256), 256, 0, stream>>>(W.f1, W.yb, 32, 32 * j, 97);
    }

    // ================= hpf =================
    h_conv(stream, W.yb, hpf_conv, 97, 97, 3, 0, W.tA);
    h_bn_nchw(stream, W.tA, hpf_bn, 97, 1, W.mv, W.tB);
    h_pack(stream, W.tB, 97, W.f0);
    h_graph(stream, W.f0, 97, W);
    cur = W.f0; alt = W.f1;
    for (int i = 0; i < 3; ++i) {
        h_gconv(stream, cur, hpf_gcs[i], 97, 97, 1, alt, W);
        float* t = cur; cur = alt; alt = t;
    }
    unpack_k<<<cdiv(tot97, 256), 256, 0, stream>>>(cur, W.yhpf, 97, 0, 97);
    add2_k<<<cdiv(tot97, 256), 256, 0, stream>>>(W.yb, W.yhpf, W.yb, tot97);

    // ================= 4x lpf: y = (y + branch(y)) + y_hpf =================
    for (int l = 0; l < 4; ++l) {
        h_conv(stream, W.yb, lpf_conv[l], 97, 97, 5, 0, W.tA);
        h_bn_nchw(stream, W.tA, lpf_bn_c[l], 97, 1, W.mv, W.tB);
        h_pack(stream, W.tB, 97, W.f0);
        h_graph(stream, W.f0, 97, W);
        cur = W.f0; alt = W.f1;
        for (int t = 0; t < 3; ++t) {
            h_gconv(stream, cur, lpf_gcs[l][t], 97, 97, 0, alt, W);
            h_bn_bnc(stream, alt, lpf_bns[l][t], 97, 1, W.mv, cur);  // result back in cur
        }
        unpack_k<<<cdiv(tot97, 256), 256, 0, stream>>>(cur, W.tA, 97, 0, 97);
        add3_k<<<cdiv(tot97, 256), 256, 0, stream>>>(W.yb, W.tA, W.yhpf, W.yb, tot97);
    }

    // ================= post: 97 -> 64 -> 32 -> 1, + x =================
    h_pack(stream, W.yb, 97, W.f0);
    h_graph(stream, W.f0, 97, W);
    h_gconv(stream, W.f0, post_gcs[0], 97, 64, 0, W.f1, W);
    h_bn_bnc(stream, W.f1, post_bns[0], 64, 1, W.mv, W.f0);

    h_graph(stream, W.f0, 64, W);
    h_gconv(stream, W.f0, post_gcs[1], 64, 32, 0, W.f1, W);
    h_bn_bnc(stream, W.f1, post_bns[1], 32, 1, W.mv, W.f0);

    h_graph(stream, W.f0, 32, W);
    h_gconv(stream, W.f0, post_gcs[2], 32, 1, 0, W.f1, W);

    final_k<<<cdiv(MROWS, 256), 256, 0, stream>>>(W.f1, x, (float*)d_out);
}
